// HyperRegionConv_79791902425269
// MI455X (gfx1250) — matrix operand order repaired and checked by reading, not compiled, not run on hardware
//
#include <hip/hip_runtime.h>
#include <hip/hip_bf16.h>

typedef __attribute__((ext_vector_type(16))) __bf16 v16bf;
typedef __attribute__((ext_vector_type(8)))  float  v8f;

#define B_     8
#define CIN    64
#define COUT   64
#define RN     4
#define H_     128
#define W_     128
#define HW     (H_ * W_)
#define KSTEPS 18        // 9 taps * 2 ic-halves of 32
#define TILE_N 64
#define BN_EPS 1e-5f

// ---------------------------------------------------------------------------
// Kernel 0: repack dynamic conv weights fp32 -> bf16 WMMA A-fragments.
// Layout: wpack[b][mblk(4)][kstep(18)][r(4)][lane(32)][e(16)]
//   -> the 4 regions' A fragments for one k-step are contiguous (4 KB),
//      so the GEMM loop can clause 8 global_load_b128 per k-step.
// Contiguous K order (matches B fragments):
//   lanes 0-15 : M = lane,    K = {0..15}
//   lanes 16-31: M = lane-16, K = {16..31}
// kstep = tap*2 + kk  (tap = ky*3+kx, kk selects ic block of 32)
// ---------------------------------------------------------------------------
__global__ void pack_weights_kernel(const float* __restrict__ kern,
                                    __bf16* __restrict__ wpack, int total) {
    int idx = blockIdx.x * 256 + threadIdx.x;
    if (idx >= total) return;
    int e     = idx & 15;
    int lane  = (idx >> 4) & 31;
    int r     = (idx >> 9) & 3;
    int t     = idx >> 11;
    int kstep = t % KSTEPS;  t /= KSTEPS;
    int mblk  = t & 3;       t >>= 2;
    int b     = t;
    int tap = kstep >> 1, kk = kstep & 1;
    int ky = tap / 3, kx = tap - 3 * ky;
    int k32 = ((lane >> 4) ? 16 : 0) + e;
    int ic  = kk * 32 + k32;
    int oc  = mblk * 16 + (lane & 15);
    size_t src = ((((size_t)(b * RN + r) * COUT + oc) * CIN + ic) * 3 + ky) * 3 + kx;
    wpack[idx] = (__bf16)kern[src];
}

// ---------------------------------------------------------------------------
// Kernel 1: region logits (3x3 conv, SAME) + bias + softmax over RN.
// Writes masks directly into the masks section of d_out.
// ---------------------------------------------------------------------------
__global__ void region_masks_kernel(const float* __restrict__ x,
                                    const float* __restrict__ mask_w,
                                    const float* __restrict__ mask_b,
                                    float* __restrict__ masks) {
    __shared__ float sw[RN * CIN * 9];
    __shared__ float sb[RN];
    for (int i = threadIdx.x; i < RN * CIN * 9; i += 256) sw[i] = mask_w[i];
    if (threadIdx.x < RN) sb[threadIdx.x] = mask_b[threadIdx.x];
    __syncthreads();

    int t = blockIdx.x * 256 + threadIdx.x;
    int b = t / HW;
    int p = t - b * HW;
    int y = p / W_, xx = p - y * W_;

    float l0 = sb[0], l1 = sb[1], l2 = sb[2], l3 = sb[3];
    const float* xb = x + (size_t)b * CIN * HW;
    for (int ic = 0; ic < CIN; ++ic) {
        const float* xp = xb + (size_t)ic * HW;
        #pragma unroll
        for (int ky = 0; ky < 3; ++ky) {
            int gy = y + ky - 1;
            if (gy < 0 || gy >= H_) continue;
            #pragma unroll
            for (int kx = 0; kx < 3; ++kx) {
                int gx = xx + kx - 1;
                if (gx < 0 || gx >= W_) continue;
                float v = xp[gy * W_ + gx];
                int wi = (ic * 3 + ky) * 3 + kx;
                l0 = fmaf(v, sw[0 * CIN * 9 + wi], l0);
                l1 = fmaf(v, sw[1 * CIN * 9 + wi], l1);
                l2 = fmaf(v, sw[2 * CIN * 9 + wi], l2);
                l3 = fmaf(v, sw[3 * CIN * 9 + wi], l3);
            }
        }
    }
    float mx = fmaxf(fmaxf(l0, l1), fmaxf(l2, l3));
    float e0 = __expf(l0 - mx), e1 = __expf(l1 - mx);
    float e2 = __expf(l2 - mx), e3 = __expf(l3 - mx);
    float inv = 1.0f / (e0 + e1 + e2 + e3);
    size_t mo = (size_t)b * RN * HW + p;
    masks[mo + 0 * HW] = e0 * inv;
    masks[mo + 1 * HW] = e1 * inv;
    masks[mo + 2 * HW] = e2 * inv;
    masks[mo + 3 * HW] = e3 * inv;
}

// ---------------------------------------------------------------------------
// Kernel 2: implicit-GEMM dynamic region conv with bf16 WMMA, fused
// mask-weighted region sum + BN + ReLU.
//
// Block = 256 threads (8 waves). Block tile: M=64 oc x N=64 pixels (half row).
// Wave tile: M=16 x N=32 (two 16x16 C tiles) x RN=4 regions kept as 8
// INDEPENDENT accumulator tiles (64 VGPRs). Loop nest: k-step outer,
// region inner. A fragments are software-pipelined one k-step ahead
// (double buffered) so the 8 WMMAs of step ks overlap the global loads
// of step ks+1 instead of stalling on s_wait_loadcnt 0.
// ---------------------------------------------------------------------------
__global__ void __launch_bounds__(256)
dyn_region_conv_kernel(const float* __restrict__ x,
                       const __bf16* __restrict__ wpack,
                       const float* __restrict__ masks,
                       const float* __restrict__ bn_gamma,
                       const float* __restrict__ bn_beta,
                       const float* __restrict__ bn_mean,
                       const float* __restrict__ bn_var,
                       float* __restrict__ out) {
    __shared__ __bf16 xs[3 * 66 * 64];   // 25,344 B
    __shared__ float  bnScale[COUT];
    __shared__ float  bnShift[COUT];

    const int tid   = threadIdx.x;
    const int b     = blockIdx.x >> 8;       // 256 tiles per batch image
    const int tile  = blockIdx.x & 255;
    const int ytile = tile >> 1;             // image row
    const int x0    = (tile & 1) * TILE_N;   // half-row start
    const int p0    = ytile * W_ + x0;

    if (tid < COUT) {
        float s = bn_gamma[tid] * rsqrtf(bn_var[tid] + BN_EPS);
        bnScale[tid] = s;
        bnShift[tid] = bn_beta[tid] - bn_mean[tid] * s;
    }

    // Stage x halo (zero-padded 'SAME' borders), fp32 -> bf16,
    // layout [row(3)][col(66)][ic(64)] so B fragments are ic-contiguous.
    const float* xb = x + (size_t)b * CIN * HW;
    for (int i = tid; i < 3 * CIN * 66; i += 256) {
        int row = i / (CIN * 66);
        int rem = i - row * (CIN * 66);
        int ic  = rem / 66;
        int col = rem - ic * 66;
        int gy = ytile + row - 1;
        int gx = x0 + col - 1;
        float v = 0.0f;
        if (gy >= 0 && gy < H_ && gx >= 0 && gx < W_)
            v = xb[(size_t)ic * HW + gy * W_ + gx];
        xs[(row * 66 + col) * 64 + ic] = (__bf16)v;
    }
    __syncthreads();

    const int wave = tid >> 5;
    const int lane = tid & 31;
    const int mblk = wave >> 1;       // which 16-oc block (0..3)
    const int nb   = wave & 1;        // which 32-pixel half (0..1)
    const int lhi  = lane >> 4;       // K-half selector for A/B fragments
    const int l15  = lane & 15;
    const int n0   = nb * 32 + l15;   // this lane's column (tile0) in [0,64)
    const int icoff = lhi * 16;       // B fragment K base within 32-ic block

    // 8 independent accumulator tiles: [region][n-tile]
    v8f c00 = {}, c01 = {}, c10 = {}, c11 = {};
    v8f c20 = {}, c21 = {}, c30 = {}, c31 = {};

    // A fragments: 4 regions contiguous at stride 512 elems per k-step.
    const __bf16* wp = wpack
        + (size_t)((b * 4 + mblk) * KSTEPS) * (RN * 512)
        + lane * 16;

    // ---- software pipeline: prime A for ks = 0 --------------------------
    v16bf a0 = *(const v16bf*)(wp + 0 * 512);
    v16bf a1 = *(const v16bf*)(wp + 1 * 512);
    v16bf a2 = *(const v16bf*)(wp + 2 * 512);
    v16bf a3 = *(const v16bf*)(wp + 3 * 512);

    #pragma unroll
    for (int ks = 0; ks < KSTEPS; ++ks) {
        const int tap = ks >> 1, kk = ks & 1;
        const int ky = tap / 3, kx = tap - 3 * ky;

        // B fragments for this k-step (shared by all 4 regions)
        const __bf16* bp0 = &xs[(ky * 66 + n0 + kx) * 64 + kk * 32 + icoff];
        v16bf bf0 = *(const v16bf*)bp0;
        v16bf bf1 = *(const v16bf*)(bp0 + 16 * 64);

        // issue next k-step's A loads before consuming this step's A
        v16bf na0 = a0, na1 = a1, na2 = a2, na3 = a3;
        if (ks + 1 < KSTEPS) {
            const __bf16* ap = wp + (size_t)(ks + 1) * (RN * 512);
            na0 = *(const v16bf*)(ap + 0 * 512);
            na1 = *(const v16bf*)(ap + 1 * 512);
            na2 = *(const v16bf*)(ap + 2 * 512);
            na3 = *(const v16bf*)(ap + 3 * 512);
        }

        c00 = __builtin_amdgcn_wmma_f32_16x16x32_bf16(false, a0, false, bf0, (short)0, c00, false, false);
        c01 = __builtin_amdgcn_wmma_f32_16x16x32_bf16(false, a0, false, bf1, (short)0, c01, false, false);
        c10 = __builtin_amdgcn_wmma_f32_16x16x32_bf16(false, a1, false, bf0, (short)0, c10, false, false);
        c11 = __builtin_amdgcn_wmma_f32_16x16x32_bf16(false, a1, false, bf1, (short)0, c11, false, false);
        c20 = __builtin_amdgcn_wmma_f32_16x16x32_bf16(false, a2, false, bf0, (short)0, c20, false, false);
        c21 = __builtin_amdgcn_wmma_f32_16x16x32_bf16(false, a2, false, bf1, (short)0, c21, false, false);
        c30 = __builtin_amdgcn_wmma_f32_16x16x32_bf16(false, a3, false, bf0, (short)0, c30, false, false);
        c31 = __builtin_amdgcn_wmma_f32_16x16x32_bf16(false, a3, false, bf1, (short)0, c31, false, false);

        a0 = na0; a1 = na1; a2 = na2; a3 = na3;
    }

    // Per-column region mask weights (one float per region per C tile).
    const float* mbase = masks + (size_t)b * RN * HW + p0;
    float m00 = mbase[0 * HW + n0], m01 = mbase[0 * HW + n0 + 16];
    float m10 = mbase[1 * HW + n0], m11 = mbase[1 * HW + n0 + 16];
    float m20 = mbase[2 * HW + n0], m21 = mbase[2 * HW + n0 + 16];
    float m30 = mbase[3 * HW + n0], m31 = mbase[3 * HW + n0 + 16];

    // Fused region sum + BatchNorm (eval) + ReLU epilogue.
    // C-tile rows: VGPR i holds M = i (lanes 0-15) or M = 8+i (lanes 16-31).
    float* ob = out + (size_t)b * COUT * HW;
    #pragma unroll
    for (int i = 0; i < 8; ++i) {
        int oc = mblk * 16 + i + lhi * 8;
        float s  = bnScale[oc];
        float sh = bnShift[oc];
        float y0 = c00[i] * m00;
        y0 = fmaf(c10[i], m10, y0);
        y0 = fmaf(c20[i], m20, y0);
        y0 = fmaf(c30[i], m30, y0);
        float y1 = c01[i] * m01;
        y1 = fmaf(c11[i], m11, y1);
        y1 = fmaf(c21[i], m21, y1);
        y1 = fmaf(c31[i], m31, y1);
        ob[(size_t)oc * HW + p0 + n0]      = fmaxf(fmaf(y0, s, sh), 0.0f);
        ob[(size_t)oc * HW + p0 + n0 + 16] = fmaxf(fmaf(y1, s, sh), 0.0f);
    }
}

// ---------------------------------------------------------------------------
// Host launcher
// ---------------------------------------------------------------------------
extern "C" void kernel_launch(void* const* d_in, const int* in_sizes, int n_in,
                              void* d_out, int out_size, void* d_ws, size_t ws_size,
                              hipStream_t stream) {
    const float* x        = (const float*)d_in[0];
    const float* kern     = (const float*)d_in[1];
    const float* mask_w   = (const float*)d_in[2];
    const float* mask_b   = (const float*)d_in[3];
    const float* bn_gamma = (const float*)d_in[4];
    const float* bn_beta  = (const float*)d_in[5];
    const float* bn_mean  = (const float*)d_in[6];
    const float* bn_var   = (const float*)d_in[7];

    float* out   = (float*)d_out;
    float* masks = out + (size_t)B_ * COUT * HW;   // tuple output #2

    __bf16* wpack = (__bf16*)d_ws;                 // 2.36 MB of scratch
    const int totalPack = B_ * 4 * KSTEPS * RN * 512;

    pack_weights_kernel<<<(totalPack + 255) / 256, 256, 0, stream>>>(
        kern, wpack, totalPack);

    region_masks_kernel<<<(B_ * HW) / 256, 256, 0, stream>>>(
        x, mask_w, mask_b, masks);

    dyn_region_conv_kernel<<<B_ * 256, 256, 0, stream>>>(
        x, wpack, masks, bn_gamma, bn_beta, bn_mean, bn_var, out);
}
